// Model_63230508532198
// MI455X (gfx1250) — matrix operand order repaired and checked by reading, not compiled, not run on hardware
//
#include <hip/hip_runtime.h>

typedef __attribute__((ext_vector_type(16))) __bf16 v16bf;
typedef __attribute__((ext_vector_type(8)))  __bf16 bf16x8;
typedef __attribute__((ext_vector_type(8)))  float  v8f;

#define S_ 4096
#define E_ 768
#define H_ 12
#define D_ 64
#define F_ 3072
#define L_ 12
#define W_ 256

static __device__ inline v8f wmma_bf16(v16bf a, v16bf b, v8f c) {
  return __builtin_amdgcn_wmma_f32_16x16x32_bf16(false, a, false, b, (short)0, c, false, false);
}
static __device__ inline v8f zero_v8f() {
  v8f v;
#pragma unroll
  for (int i = 0; i < 8; ++i) v[i] = 0.0f;
  return v;
}
union V16U { v16bf v; bf16x8 h[2]; };

// A-fragment (16x32 bf16, MxK): lane = m (0..15) in both halves; K in
// [2*khalf, 2*khalf+16) with khalf = 8*(lane>=16).
static __device__ inline v16bf load_afrag(const __bf16* p, int khalf) {
  V16U u;
  u.h[0] = *(const bf16x8*)(p + 2 * khalf);
  u.h[1] = *(const bf16x8*)(p + 2 * khalf + 8);
  return u.v;
}
// 16 contiguous bf16 (one lane's B-fragment slice)
static __device__ inline v16bf load_b16(const __bf16* p) {
  V16U u;
  u.h[0] = *(const bf16x8*)(p);
  u.h[1] = *(const bf16x8*)(p + 8);
  return u.v;
}

// Async 16B copy global -> LDS (GVS mode: mem = saddr + vaddr32). Tracked by
// ASYNCcnt. ldsoff is a byte offset into the kernel's LDS allocation (the
// single __shared__ array sits at offset 0).
static __device__ inline void async_copy_b128(unsigned ldsoff, unsigned goff,
                                              const void* sbase) {
  asm volatile("global_load_async_to_lds_b128 %0, %1, %2"
               :: "v"(ldsoff), "v"(goff), "s"(sbase)
               : "memory");
}
template <int N>
static __device__ inline void wait_asynccnt() {
  asm volatile("s_wait_asynccnt %0" :: "i"(N) : "memory");
}

// ---------------------------------------------------------------------------
// Weight conversion: fp32 row-major [K,N] -> bf16 pre-swizzled WMMA B tiles.
// Tile (kt,nt) covers K rows [kt*32,kt*32+32), cols [nt*16,nt*16+16).
// Storage: ((kt*(N/16)+nt)*32 + lane)*16 + e, where lane: n=lane&15,
// kbase=(lane>>4)*16, element e -> K = kt*32 + kbase + e.
// ---------------------------------------------------------------------------
__global__ __launch_bounds__(256) void convert_b_kernel(
    const float* __restrict__ B, __bf16* __restrict__ Bt, int K, int N) {
  int idx  = blockIdx.x * 256 + threadIdx.x;
  int lane = idx & 31;
  int t    = idx >> 5;
  int ntile16 = N >> 4;
  int total_t = (K >> 5) * ntile16;
  if (t >= total_t) return;
  int kt = t / ntile16, nt = t % ntile16;
  int nn = lane & 15, kb = (lane >> 4) * 16;
  const float* src = B + (size_t)(kt * 32 + kb) * N + nt * 16 + nn;
  __bf16* dst = Bt + (((size_t)t * 32 + lane) << 4);
#pragma unroll
  for (int e = 0; e < 16; ++e) dst[e] = (__bf16)src[(size_t)e * N];
}

// ---------------------------------------------------------------------------
// bf16 WMMA GEMM: C[M,N] = A[M,K] (row-major bf16) x Bt (pre-swizzled bf16).
// Block tile 256 rows x 64 cols; wave tile 32x64 (8 accumulators).
// Per K-step (32), the 4KB B slab (4 consecutive n-tiles, contiguous in Bt)
// is async-copied into a double-buffered LDS stage shared by all 8 waves:
// each of the 256 threads moves one 16-byte chunk via
// global_load_async_to_lds_b128; sync = s_wait_asynccnt + s_barrier.
// EPI 1: +bias, *scale -> qkv [head][row][64] bf16
// EPI 2: +bias -> vT [head][d][row] bf16 (transposed for attention B-frags)
// EPI 3: +bias, exact GELU -> bf16 [M,N]
// EPI 4: +bias, +res (f32 [M,N]) -> f32 [M,N]
// ---------------------------------------------------------------------------
template <int EPI>
__global__ __launch_bounds__(256) void gemm_wmma_bf16(
    const __bf16* __restrict__ A, const __bf16* __restrict__ Bt,
    const float* __restrict__ bias, const float* __restrict__ res,
    float* __restrict__ outf, __bf16* __restrict__ outb,
    int M, int N, int K, float scale) {
  __shared__ __align__(16) __bf16 Bsh[2 * 2048];  // 2 x 4KB double buffer
  int tid  = threadIdx.x;
  int lane = tid & 31, wid = tid >> 5;
  int nbx  = N >> 6;
  int bm = blockIdx.x / nbx, bn = blockIdx.x % nbx;
  int row0 = bm * 256 + wid * 32;
  int col0 = bn << 6;
  int ln15 = lane & 15, half = lane >> 4, khalf = half * 8;
  int ntile16 = N >> 4;
  int nt0 = col0 >> 4;

  v8f acc[2][4];
#pragma unroll
  for (int i = 0; i < 2; ++i)
#pragma unroll
    for (int t = 0; t < 4; ++t) acc[i][t] = zero_v8f();

  const __bf16* arow0 = A + (size_t)(row0 + ln15) * K;
  const __bf16* arow1 = A + (size_t)(row0 + 16 + ln15) * K;

  const int nk = K >> 5;
  unsigned thr16 = (unsigned)tid * 16u;  // this thread's 16B chunk
  // prologue: stage tile kt=0 into buffer 0
  async_copy_b128(thr16, (unsigned)((0u * ntile16 + nt0) * 1024u) + thr16, Bt);

  for (int kt = 0; kt < nk; ++kt) {
    if (kt + 1 < nk) {
      // prefetch next slab into the other buffer (safe: barrier at end of the
      // previous iteration guarantees no wave still reads it)
      async_copy_b128(((unsigned)((kt + 1) & 1) * 4096u) + thr16,
                      (unsigned)(((unsigned)(kt + 1) * ntile16 + nt0) * 1024u) + thr16,
                      Bt);
      wait_asynccnt<1>();   // current slab (issued one iteration ago) done
    } else {
      wait_asynccnt<0>();
    }
    __syncthreads();        // all waves' copies of current slab visible

    const __bf16* bb = &Bsh[(kt & 1) * 2048];
    v16bf a0 = load_afrag(arow0 + kt * 32, khalf);
    v16bf a1 = load_afrag(arow1 + kt * 32, khalf);
#pragma unroll
    for (int t = 0; t < 4; ++t) {
      v16bf b = load_b16(bb + ((t * 32 + lane) << 4));
      acc[0][t] = wmma_bf16(a0, b, acc[0][t]);
      acc[1][t] = wmma_bf16(a1, b, acc[1][t]);
    }
    __syncthreads();        // everyone done reading before it is overwritten
  }

#pragma unroll
  for (int i = 0; i < 2; ++i) {
#pragma unroll
    for (int t = 0; t < 4; ++t) {
      int col = col0 + t * 16 + ln15;      // C layout: lane&15 = column
      float bcol = bias[col];
#pragma unroll
      for (int r = 0; r < 8; ++r) {
        int row = row0 + i * 16 + r + 8 * half;  // C rows: r + 8*half
        float val = acc[i][t][r] + bcol;
        if (EPI == 1) {
          int head = col >> 6, d = col & 63;
          outb[(((size_t)head * S_ + row) << 6) + d] = (__bf16)(val * scale);
        } else if (EPI == 2) {
          int head = col >> 6, d = col & 63;
          outb[(size_t)(head * 64 + d) * S_ + row] = (__bf16)val;
        } else if (EPI == 3) {
          float gl = 0.5f * val * (1.0f + erff(val * 0.70710678118654752f));
          outb[(size_t)row * N + col] = (__bf16)gl;
        } else {
          outf[(size_t)row * N + col] = val + res[(size_t)row * N + col];
        }
      }
    }
  }
}

// ---------------------------------------------------------------------------
// Banded (sliding-window) attention, flash-style. One wave = 16 query rows.
// q pre-scaled by 1/sqrt(D). k: [head][row][64]; v transposed: [head][d][row].
// Streams 24 chunks of 32 keys over [qb*W - W, qb*W + 2W).
// ---------------------------------------------------------------------------
__global__ __launch_bounds__(256) void attn_kernel(
    const __bf16* __restrict__ qh, const __bf16* __restrict__ kh,
    const __bf16* __restrict__ vT, __bf16* __restrict__ attnb) {
  __shared__ __align__(16) __bf16 Psh[8 * 512];
  int wid  = threadIdx.x >> 5;
  int lane = threadIdx.x & 31;
  int wave = blockIdx.x * 8 + wid;
  const int tiles = S_ / 16;
  if (wave >= H_ * tiles) return;
  int h  = wave / tiles;
  int i0 = (wave % tiles) << 4;
  int qb = i0 / W_;
  int jstart = qb * W_ - W_;
  int ln15 = lane & 15, half = lane >> 4;
  int khalf = half * 8, kbase = half * 16;

  const __bf16* qrow = qh + (((size_t)h * S_ + i0 + ln15) << 6);
  v16bf aq0 = load_afrag(qrow, khalf);        // d = 0..31
  v16bf aq1 = load_afrag(qrow + 32, khalf);   // d = 32..63

  v8f Oacc[4];
#pragma unroll
  for (int t = 0; t < 4; ++t) Oacc[t] = zero_v8f();
  float mrow[8], lrow[8];
#pragma unroll
  for (int r = 0; r < 8; ++r) { mrow[r] = -1e30f; lrow[r] = 0.0f; }

  __bf16* P = Psh + wid * 512;

  for (int c = 0; c < 24; ++c) {
    int jc = jstart + c * 32;
    if (jc < 0 || jc >= S_) continue;

    // scores: two 16x16 tiles over key columns [jc, jc+16) and [jc+16, jc+32)
    const __bf16* krow0 = kh + (((size_t)h * S_ + jc + ln15) << 6);
    const __bf16* krow1 = kh + (((size_t)h * S_ + jc + 16 + ln15) << 6);
    v8f s0 = zero_v8f(), s1 = zero_v8f();
    s0 = wmma_bf16(aq0, load_b16(krow0 + kbase), s0);
    s0 = wmma_bf16(aq1, load_b16(krow0 + 32 + kbase), s0);
    s1 = wmma_bf16(aq0, load_b16(krow1 + kbase), s1);
    s1 = wmma_bf16(aq1, load_b16(krow1 + 32 + kbase), s1);

    float alpha[8];
#pragma unroll
    for (int r = 0; r < 8; ++r) {
      int row = i0 + r + 8 * half;
      int dj0 = (jc + ln15) - row;
      int dj1 = (jc + 16 + ln15) - row;
      if (dj0 < -W_ || dj0 > W_) s0[r] = -1e9f;
      if (dj1 < -W_ || dj1 > W_) s1[r] = -1e9f;
      float cm = fmaxf(s0[r], s1[r]);
#pragma unroll
      for (int off = 1; off < 16; off <<= 1) cm = fmaxf(cm, __shfl_xor(cm, off, 32));
      float mnew = fmaxf(mrow[r], cm);
      alpha[r] = __expf(mrow[r] - mnew);
      mrow[r] = mnew;
      float p0 = __expf(s0[r] - mnew);
      float p1 = __expf(s1[r] - mnew);
      float cs = p0 + p1;
#pragma unroll
      for (int off = 1; off < 16; off <<= 1) cs += __shfl_xor(cs, off, 32);
      lrow[r] = lrow[r] * alpha[r] + cs;
      P[(r + 8 * half) * 32 + ln15]      = (__bf16)p0;
      P[(r + 8 * half) * 32 + 16 + ln15] = (__bf16)p1;
    }
#pragma unroll
    for (int t = 0; t < 4; ++t)
#pragma unroll
      for (int r = 0; r < 8; ++r) Oacc[t][r] *= alpha[r];

    // wave-local LDS RAW: DS ops are in-order per wave; keep compiler honest.
    asm volatile("s_wait_dscnt 0x0" ::: "memory");
    v16bf pa = load_afrag(P + ln15 * 32, khalf);   // P as A-frag (16x32)
#pragma unroll
    for (int t = 0; t < 4; ++t) {
      const __bf16* vrow =
          vT + (size_t)(h * 64 + t * 16 + ln15) * S_ + jc + kbase;
      Oacc[t] = wmma_bf16(pa, load_b16(vrow), Oacc[t]);
    }
  }

#pragma unroll
  for (int r = 0; r < 8; ++r) {
    float inv = 1.0f / lrow[r];
    int row = i0 + r + 8 * half;
#pragma unroll
    for (int t = 0; t < 4; ++t)
      attnb[(size_t)row * E_ + h * 64 + t * 16 + ln15] =
          (__bf16)(Oacc[t][r] * inv);
  }
}

// ---------------------------------------------------------------------------
// LayerNorm over E=768, one block per row, 256 threads x 3 elements.
// ---------------------------------------------------------------------------
__global__ __launch_bounds__(256) void ln_kernel(
    const float* __restrict__ in, const float* __restrict__ g,
    const float* __restrict__ b, float* __restrict__ outf,
    __bf16* __restrict__ outb) {
  __shared__ float red[256];
  int row = blockIdx.x, tid = threadIdx.x;
  const float* x = in + (size_t)row * E_;
  float v0 = x[tid], v1 = x[tid + 256], v2 = x[tid + 512];
  red[tid] = v0 + v1 + v2;
  __syncthreads();
  for (int st = 128; st > 0; st >>= 1) {
    if (tid < st) red[tid] += red[tid + st];
    __syncthreads();
  }
  float mean = red[0] * (1.0f / 768.0f);
  __syncthreads();
  float d0 = v0 - mean, d1 = v1 - mean, d2 = v2 - mean;
  red[tid] = d0 * d0 + d1 * d1 + d2 * d2;
  __syncthreads();
  for (int st = 128; st > 0; st >>= 1) {
    if (tid < st) red[tid] += red[tid + st];
    __syncthreads();
  }
  float rstd = rsqrtf(red[0] * (1.0f / 768.0f) + 1e-5f);
  float y0 = d0 * rstd * g[tid] + b[tid];
  float y1 = d1 * rstd * g[tid + 256] + b[tid + 256];
  float y2 = d2 * rstd * g[tid + 512] + b[tid + 512];
  size_t o = (size_t)row * E_;
  outf[o + tid] = y0;        outb[o + tid] = (__bf16)y0;
  outf[o + tid + 256] = y1;  outb[o + tid + 256] = (__bf16)y1;
  outf[o + tid + 512] = y2;  outb[o + tid + 512] = (__bf16)y2;
}

// Embedding gather + LayerNorm (same reduction structure).
__global__ __launch_bounds__(256) void embed_ln_kernel(
    const int* __restrict__ x, const float* __restrict__ we,
    const float* __restrict__ pe, const float* __restrict__ g,
    const float* __restrict__ b, float* __restrict__ outf,
    __bf16* __restrict__ outb) {
  __shared__ float red[256];
  int row = blockIdx.x, tid = threadIdx.x;
  int tok = x[row];
  const float* wr = we + (size_t)tok * E_;
  const float* pr = pe + (size_t)row * E_;
  float v0 = wr[tid] + pr[tid];
  float v1 = wr[tid + 256] + pr[tid + 256];
  float v2 = wr[tid + 512] + pr[tid + 512];
  red[tid] = v0 + v1 + v2;
  __syncthreads();
  for (int st = 128; st > 0; st >>= 1) {
    if (tid < st) red[tid] += red[tid + st];
    __syncthreads();
  }
  float mean = red[0] * (1.0f / 768.0f);
  __syncthreads();
  float d0 = v0 - mean, d1 = v1 - mean, d2 = v2 - mean;
  red[tid] = d0 * d0 + d1 * d1 + d2 * d2;
  __syncthreads();
  for (int st = 128; st > 0; st >>= 1) {
    if (tid < st) red[tid] += red[tid + st];
    __syncthreads();
  }
  float rstd = rsqrtf(red[0] * (1.0f / 768.0f) + 1e-5f);
  float y0 = d0 * rstd * g[tid] + b[tid];
  float y1 = d1 * rstd * g[tid + 256] + b[tid + 256];
  float y2 = d2 * rstd * g[tid + 512] + b[tid + 512];
  size_t o = (size_t)row * E_;
  outf[o + tid] = y0;        outb[o + tid] = (__bf16)y0;
  outf[o + tid + 256] = y1;  outb[o + tid + 256] = (__bf16)y1;
  outf[o + tid + 512] = y2;  outb[o + tid + 512] = (__bf16)y2;
}

__global__ void copy_row0_kernel(const float* __restrict__ h,
                                 float* __restrict__ dst) {
  int i = blockIdx.x * 256 + threadIdx.x;
  if (i < E_) dst[i] = h[i];
}

// cls = mean over 5 saved CLS rows; out = cls @ w_head + b_head
__global__ __launch_bounds__(256) void head_kernel(
    const float* __restrict__ cls, const float* __restrict__ wh,
    const float* __restrict__ bh, float* __restrict__ out) {
  __shared__ float red[256];
  int tid = threadIdx.x;
  float s = 0.0f;
  for (int i = tid; i < E_; i += 256) {
    float m = 0.0f;
#pragma unroll
    for (int l = 0; l < 5; ++l) m += cls[l * E_ + i];
    s += (m * 0.2f) * wh[i];
  }
  red[tid] = s;
  __syncthreads();
  for (int st = 128; st > 0; st >>= 1) {
    if (tid < st) red[tid] += red[tid + st];
    __syncthreads();
  }
  if (tid == 0) out[0] = red[0] + bh[0];
}

// ---------------------------------------------------------------------------
extern "C" void kernel_launch(void* const* d_in, const int* in_sizes, int n_in,
                              void* d_out, int out_size, void* d_ws,
                              size_t ws_size, hipStream_t stream) {
  (void)in_sizes; (void)n_in; (void)out_size; (void)ws_size;
  const int*   x    = (const int*)d_in[0];
  const float* we   = (const float*)d_in[1];
  const float* pe   = (const float*)d_in[2];
  const float* lnes = (const float*)d_in[3];
  const float* lneb = (const float*)d_in[4];
  const float* Wq   = (const float*)d_in[5];
  const float* bq   = (const float*)d_in[6];
  const float* Wk   = (const float*)d_in[7];
  const float* bk   = (const float*)d_in[8];
  const float* Wv   = (const float*)d_in[9];
  const float* bv   = (const float*)d_in[10];
  const float* Wo   = (const float*)d_in[11];
  const float* bo   = (const float*)d_in[12];
  const float* ln1s = (const float*)d_in[13];
  const float* ln1b = (const float*)d_in[14];
  const float* W1   = (const float*)d_in[15];
  const float* b1   = (const float*)d_in[16];
  const float* W2   = (const float*)d_in[17];
  const float* b2   = (const float*)d_in[18];
  const float* ln2s = (const float*)d_in[19];
  const float* ln2b = (const float*)d_in[20];
  const float* wh   = (const float*)d_in[21];
  const float* bh   = (const float*)d_in[22];
  float* out = (float*)d_out;

  char* base = (char*)d_ws;
  size_t off = 0;
  auto alloc = [&](size_t bytes) -> char* {
    char* p = base + off;
    off += (bytes + 255) & ~(size_t)255;
    return p;
  };
  float*  h     = (float*)alloc((size_t)S_ * E_ * 4);
  float*  t1    = (float*)alloc((size_t)S_ * E_ * 4);
  float*  h1    = (float*)alloc((size_t)S_ * E_ * 4);
  __bf16* hb    = (__bf16*)alloc((size_t)S_ * E_ * 2);
  __bf16* h1b   = (__bf16*)alloc((size_t)S_ * E_ * 2);
  __bf16* qh    = (__bf16*)alloc((size_t)S_ * E_ * 2);
  __bf16* kk    = (__bf16*)alloc((size_t)S_ * E_ * 2);
  __bf16* vT    = (__bf16*)alloc((size_t)S_ * E_ * 2);
  __bf16* attnb = (__bf16*)alloc((size_t)S_ * E_ * 2);
  __bf16* f1b   = (__bf16*)alloc((size_t)S_ * F_ * 2);
  __bf16* wq_t  = (__bf16*)alloc((size_t)E_ * E_ * 2);
  __bf16* wk_t  = (__bf16*)alloc((size_t)E_ * E_ * 2);
  __bf16* wv_t  = (__bf16*)alloc((size_t)E_ * E_ * 2);
  __bf16* wo_t  = (__bf16*)alloc((size_t)E_ * E_ * 2);
  __bf16* w1_t  = (__bf16*)alloc((size_t)E_ * F_ * 2);
  __bf16* w2_t  = (__bf16*)alloc((size_t)F_ * E_ * 2);
  float*  cls   = (float*)alloc((size_t)5 * E_ * 4);

  embed_ln_kernel<<<S_, 256, 0, stream>>>(x, we, pe, lnes, lneb, h, hb);

  const int cb_ee  = (E_ * E_ / 16) / 256;            // 144
  const int cb_ef  = (E_ * F_ / 16) / 256;            // 576
  const int gb768  = (S_ / 256) * (E_ / 64);          // 192 (block 256x64)
  const int gb3072 = (S_ / 256) * (F_ / 64);          // 768
  const int ab     = (H_ * (S_ / 16)) / 8;            // 384

  for (int l = 0; l < L_; ++l) {
    const float* wql = Wq + (size_t)l * E_ * E_;
    const float* wkl = Wk + (size_t)l * E_ * E_;
    const float* wvl = Wv + (size_t)l * E_ * E_;
    const float* wol = Wo + (size_t)l * E_ * E_;
    const float* w1l = W1 + (size_t)l * E_ * F_;
    const float* w2l = W2 + (size_t)l * F_ * E_;

    convert_b_kernel<<<cb_ee, 256, 0, stream>>>(wql, wq_t, E_, E_);
    convert_b_kernel<<<cb_ee, 256, 0, stream>>>(wkl, wk_t, E_, E_);
    convert_b_kernel<<<cb_ee, 256, 0, stream>>>(wvl, wv_t, E_, E_);
    convert_b_kernel<<<cb_ee, 256, 0, stream>>>(wol, wo_t, E_, E_);
    convert_b_kernel<<<cb_ef, 256, 0, stream>>>(w1l, w1_t, E_, F_);
    convert_b_kernel<<<cb_ef, 256, 0, stream>>>(w2l, w2_t, F_, E_);

    gemm_wmma_bf16<1><<<gb768, 256, 0, stream>>>(
        hb, wq_t, bq + (size_t)l * E_, nullptr, nullptr, qh, S_, E_, E_, 0.125f);
    gemm_wmma_bf16<1><<<gb768, 256, 0, stream>>>(
        hb, wk_t, bk + (size_t)l * E_, nullptr, nullptr, kk, S_, E_, E_, 1.0f);
    gemm_wmma_bf16<2><<<gb768, 256, 0, stream>>>(
        hb, wv_t, bv + (size_t)l * E_, nullptr, nullptr, vT, S_, E_, E_, 1.0f);

    attn_kernel<<<ab, 256, 0, stream>>>(qh, kk, vT, attnb);

    gemm_wmma_bf16<4><<<gb768, 256, 0, stream>>>(
        attnb, wo_t, bo + (size_t)l * E_, h, t1, nullptr, S_, E_, E_, 1.0f);
    ln_kernel<<<S_, 256, 0, stream>>>(t1, ln1s + (size_t)l * E_,
                                      ln1b + (size_t)l * E_, h1, h1b);

    gemm_wmma_bf16<3><<<gb3072, 256, 0, stream>>>(
        h1b, w1_t, b1 + (size_t)l * F_, nullptr, nullptr, f1b, S_, F_, E_, 1.0f);
    gemm_wmma_bf16<4><<<gb768, 256, 0, stream>>>(
        f1b, w2_t, b2 + (size_t)l * E_, h1, t1, nullptr, S_, E_, F_, 1.0f);
    ln_kernel<<<S_, 256, 0, stream>>>(t1, ln2s + (size_t)l * E_,
                                      ln2b + (size_t)l * E_, h, hb);

    if (l >= L_ - 5)
      copy_row0_kernel<<<3, 256, 0, stream>>>(h, cls + (size_t)(l - (L_ - 5)) * E_);
  }

  head_kernel<<<1, 256, 0, stream>>>(cls, wh, bh, out);
}
